// TransformerDecoderLayerWithMixedAttention_65077344469581
// MI455X (gfx1250) — matrix operand order repaired and checked by reading, not compiled, not run on hardware
//
#include <hip/hip_runtime.h>
#include <hip/hip_bf16.h>
#include <math.h>

// ---------------------------------------------------------------------------
// Types for CDNA5 WMMA (wave32): 16x16x32 bf16 -> f32
// ---------------------------------------------------------------------------
typedef __bf16 bf16_t;
typedef __attribute__((ext_vector_type(16))) __bf16 v16bf;
typedef __attribute__((ext_vector_type(8)))  __bf16 v8bf;
typedef __attribute__((ext_vector_type(8)))  float  v8f;
typedef __attribute__((ext_vector_type(4)))  unsigned int v4u;
typedef __attribute__((ext_vector_type(8)))  int v8i_t;
typedef __attribute__((ext_vector_type(4)))  int v4i_t;

#define DEV __device__ __forceinline__

DEV v8f wmma_bf16(v16bf a, v16bf b, v8f c) {
  // (neg_a, A, neg_b, B, c_mod, C, reuse_a, reuse_b)
  return __builtin_amdgcn_wmma_f32_16x16x32_bf16(false, a, false, b, (short)0, c,
                                                 false, false);
}

// A-matrix 16x32 bf16 fragment: lane L: row m = L&15, khalf = L>>4
// covers k = khalf*16 .. khalf*16+15 contiguous -> two 16B loads.
DEV v16bf afrag_bf16(const bf16_t* row, int kb, int khalf) {
  v8bf lo = *(const v8bf*)(row + kb + (khalf << 4));
  v8bf hi = *(const v8bf*)(row + kb + (khalf << 4) + 8);
  return __builtin_shufflevector(lo, hi, 0, 1, 2, 3, 4, 5, 6, 7,
                                 8, 9, 10, 11, 12, 13, 14, 15);
}

DEV v16bf afrag_f32(const float* row, int kb, int khalf) {
  const float* p0 = row + kb + (khalf << 4);
  const float* p1 = row + kb + (khalf << 4) + 8;
  v16bf f;
#pragma unroll
  for (int e = 0; e < 8; ++e) {
    f[e]     = (bf16_t)p0[e];
    f[e + 8] = (bf16_t)p1[e];
  }
  return f;
}

// B-matrix 32x16 fragment: lane L: col n = L&15, khalf = L>>4,
// k = khalf*16 .. khalf*16+15 contiguous per lane.
DEV v16bf bfrag_bf16(const bf16_t* row, int kb, int khalf) {
  const v8bf* p = (const v8bf*)(row + kb + (khalf << 4));
  v8bf lo = p[0], hi = p[1];
  return __builtin_shufflevector(lo, hi, 0, 1, 2, 3, 4, 5, 6, 7,
                                 8, 9, 10, 11, 12, 13, 14, 15);
}

DEV v16bf bfrag_f32(const float* row, int kb, int khalf) {
  const float* p0 = row + kb + (khalf << 4);
  v16bf f;
#pragma unroll
  for (int e = 0; e < 16; ++e) f[e] = (bf16_t)p0[e];
  return f;
}

// ---------------------------------------------------------------------------
// Tensor Data Mover: 2D tile Global -> LDS (D# per ISA cdna5 ch.8)
// This toolchain's builtin takes 6 args:
//   (uint32x4 g0, int32x8 g1, int32x4 g2, int32x4 g3, int32x8 extra, i32 cpol)
// ---------------------------------------------------------------------------
#if __has_builtin(__builtin_amdgcn_tensor_load_to_lds)
#define HAVE_TDM 1
DEV void tdm_load_2d(unsigned int lds_off, const void* gptr,
                     unsigned int tile_w, unsigned int tile_h,
                     unsigned long long row_stride_elems,
                     unsigned int tensor_w, unsigned int tensor_h) {
  unsigned long long ga = (unsigned long long)(uintptr_t)gptr;
  v4u g0;
  g0[0] = 1u;                                            // count=1, user mode
  g0[1] = lds_off;                                       // lds_addr (bytes)
  g0[2] = (unsigned int)ga;                              // global_addr[31:0]
  g0[3] = (unsigned int)((ga >> 32) & 0x01FFFFFFu) | (2u << 30);  // [56:32], type=2
  v8i_t g1;
  g1[0] = (int)(2u << 16);                               // wg_mask=0, data_size=4B
  g1[1] = (int)((tensor_w & 0xFFFFu) << 16);             // tensor_dim0 lo16
  g1[2] = (int)((tensor_w >> 16) | ((tensor_h & 0xFFFFu) << 16));
  g1[3] = (int)((tensor_h >> 16) | (tile_w << 16));      // tile_dim0
  g1[4] = (int)(tile_h & 0xFFFFu);                       // tile_dim1 (tile_dim2=0)
  g1[5] = (int)(unsigned int)(row_stride_elems & 0xFFFFFFFFu); // dim0_stride lo
  g1[6] = (int)(unsigned int)((row_stride_elems >> 32) & 0xFFFFu);
  g1[7] = 0;
  v4i_t z4 = {0, 0, 0, 0};
  v8i_t z8 = {0, 0, 0, 0, 0, 0, 0, 0};
  __builtin_amdgcn_tensor_load_to_lds(g0, g1, z4, z4, z8, 0);
}
#else
#define HAVE_TDM 0
#endif

// ---------------------------------------------------------------------------
// f32 -> bf16 elementwise convert
// ---------------------------------------------------------------------------
__global__ void cvt_f32_bf16_kernel(const float* __restrict__ in,
                                    bf16_t* __restrict__ out, size_t n) {
  size_t stride = (size_t)gridDim.x * blockDim.x;
  for (size_t i = (size_t)blockIdx.x * blockDim.x + threadIdx.x; i < n; i += stride)
    out[i] = (bf16_t)in[i];
}

// ---------------------------------------------------------------------------
// GEMM:  C[M,N] = A[M,K] (bf16) @ W[N,K]^T (bf16) + bias[N], optional ReLU.
// 256 threads = 8 waves; each wave owns a 16(M) x 64(N) strip (A-frag reuse
// across 4 B-frags -> 4 wmma per k-step). Block tile: 64(M) x 128(N).
// ---------------------------------------------------------------------------
__global__ void __launch_bounds__(256)
gemm_bf16_kernel(const bf16_t* __restrict__ A, const bf16_t* __restrict__ Wt,
                 const float* __restrict__ bias, float* __restrict__ C,
                 int M, int N, int K, int relu) {
  int lane  = threadIdx.x & 31;
  int wid   = threadIdx.x >> 5;
  int mt    = wid & 3;               // 4 m strips
  int nt    = wid >> 2;              // 2 n strips
  int m0    = blockIdx.y * 64 + mt * 16;
  int n0    = blockIdx.x * 128 + nt * 64;
  int n     = lane & 15;
  int khalf = lane >> 4;

  const bf16_t* arow = A + (size_t)(m0 + n) * K;
  const bf16_t* brow[4];
#pragma unroll
  for (int t = 0; t < 4; ++t) brow[t] = Wt + (size_t)(n0 + t * 16 + n) * K;

  v8f acc[4];
#pragma unroll
  for (int t = 0; t < 4; ++t) acc[t] = (v8f){};

  for (int kb = 0; kb < K; kb += 32) {
    __builtin_prefetch(arow + kb + 256, 0, 1);
    __builtin_prefetch(brow[0] + kb + 256, 0, 1);
    v16bf af = afrag_bf16(arow, kb, khalf);
#pragma unroll
    for (int t = 0; t < 4; ++t) {
      v16bf bfrg = bfrag_bf16(brow[t], kb, khalf);
      acc[t] = wmma_bf16(af, bfrg, acc[t]);
    }
  }

#pragma unroll
  for (int t = 0; t < 4; ++t) {
    float bv = bias ? bias[n0 + t * 16 + n] : 0.0f;
#pragma unroll
    for (int r = 0; r < 8; ++r) {
      float v = acc[t][r] + bv;
      if (relu) v = fmaxf(v, 0.0f);
      C[(size_t)(m0 + r + (khalf << 3)) * N + (n0 + t * 16 + n)] = v;
    }
  }
}

// ---------------------------------------------------------------------------
// Flash attention with banded+strided mask.
// Q/K/V layout [S*B, H*64], row = s*B + b.  4 waves/block share one (b,h):
// K/V 32x64 tiles staged in LDS once per block via the Tensor Data Mover.
// ---------------------------------------------------------------------------
__global__ void __launch_bounds__(128)
flash_mask_attn_kernel(const float* __restrict__ Qp, const float* __restrict__ Kp,
                       const float* __restrict__ Vp, float* __restrict__ Op,
                       int S, int Bn, int Hn, int Wband) {
  const int hd   = 64;
  int lane  = threadIdx.x & 31;
  int wid   = threadIdx.x >> 5;
  int qtile = blockIdx.x * 4 + wid;
  int bh    = blockIdx.y;
  int b     = bh % Bn;
  int h     = bh / Bn;
  int qbase = qtile * 16;
  int ld    = Hn * hd;             // 512
  int colh  = h * hd;
  int n     = lane & 15;
  int khalf = lane >> 4;
  const float inv_sqrt = 0.125f;   // 1/sqrt(64)
  const int stride2w = 2 * Wband;  // 10

  __shared__ float  Kt[32 * 64];
  __shared__ float  Vt[32 * 64];
  __shared__ __bf16 Plds[4][16 * 32];
  __bf16* pl = Plds[wid];

  // Q fragments (held in registers for the whole k-loop)
  const float* qrow = Qp + ((size_t)(qbase + n) * Bn + b) * ld + colh;
  v16bf qf0 = afrag_f32(qrow, 0, khalf);
  v16bf qf1 = afrag_f32(qrow, 32, khalf);

  v8f o[4];
  float m_run[8], l_run[8];
#pragma unroll
  for (int dc = 0; dc < 4; ++dc) o[dc] = (v8f){};
#pragma unroll
  for (int r = 0; r < 8; ++r) { m_run[r] = -__builtin_inff(); l_run[r] = 0.0f; }

  int njt = S / 32;
  for (int jt = 0; jt < njt; ++jt) {
    int jb = jt * 32;
    const float* gk = Kp + ((size_t)jb * Bn + b) * ld + colh;
    const float* gv = Vp + ((size_t)jb * Bn + b) * ld + colh;

    __syncthreads();   // previous tile fully consumed before overwrite
#if HAVE_TDM
    if (wid == 0) {
      tdm_load_2d((unsigned int)(uintptr_t)Kt, gk, 64, 32,
                  (unsigned long long)Bn * ld, 64, 32);
      tdm_load_2d((unsigned int)(uintptr_t)Vt, gv, 64, 32,
                  (unsigned long long)Bn * ld, 64, 32);
      __builtin_amdgcn_s_wait_tensorcnt(0);
    }
#else
    for (int idx = threadIdx.x; idx < 32 * 64; idx += 128) {
      int rr = idx >> 6, cc = idx & 63;
      size_t go = ((size_t)(jb + rr) * Bn + b) * ld + colh + cc;
      Kt[idx] = Kp[go];
      Vt[idx] = Vp[go];
    }
#endif
    __syncthreads();

    // ---- scores: 16(q) x 32(j) as two 16x16 WMMA tiles (K/V from LDS) ----
    const float* kl0 = Kt + (size_t)n * 64;          // j = jb + n
    const float* kl1 = Kt + (size_t)(16 + n) * 64;   // j = jb + 16 + n
    v8f s0 = {}, s1 = {};
    s0 = wmma_bf16(qf0, bfrag_f32(kl0, 0, khalf), s0);
    s0 = wmma_bf16(qf1, bfrag_f32(kl0, 32, khalf), s0);
    s1 = wmma_bf16(qf0, bfrag_f32(kl1, 0, khalf), s1);
    s1 = wmma_bf16(qf1, bfrag_f32(kl1, 32, khalf), s1);

    // ---- mask + online softmax (rows live across 16 lanes) ---------------
#pragma unroll
    for (int r = 0; r < 8; ++r) {
      int i  = qbase + r + (khalf << 3);
      int j0 = jb + n;
      int j1 = jb + 16 + n;
      bool al0 = ((j0 >= i - Wband) && (j0 <= i + Wband)) || ((j0 % stride2w) == 0);
      bool al1 = ((j1 >= i - Wband) && (j1 <= i + Wband)) || ((j1 % stride2w) == 0);
      float a0 = al0 ? s0[r] * inv_sqrt : -__builtin_inff();
      float a1 = al1 ? s1[r] * inv_sqrt : -__builtin_inff();

      float rm = fmaxf(a0, a1);
#pragma unroll
      for (int msk = 1; msk < 16; msk <<= 1) rm = fmaxf(rm, __shfl_xor(rm, msk, 32));
      float mnew = fmaxf(m_run[r], rm);
      float sc   = __expf(m_run[r] - mnew);
      float p0v  = __expf(a0 - mnew);
      float p1v  = __expf(a1 - mnew);
      float rs = p0v + p1v;
#pragma unroll
      for (int msk = 1; msk < 16; msk <<= 1) rs += __shfl_xor(rs, msk, 32);
      l_run[r] = l_run[r] * sc + rs;
      m_run[r] = mnew;
#pragma unroll
      for (int dc = 0; dc < 4; ++dc) o[dc][r] *= sc;

      // stage P (C-layout -> row-major bf16 in LDS)
      pl[(r + (khalf << 3)) * 32 + n]      = (bf16_t)p0v;
      pl[(r + (khalf << 3)) * 32 + 16 + n] = (bf16_t)p1v;
    }

    // same-wave LDS RAW: wait for ds stores before fragment reload
    asm volatile("s_wait_dscnt 0" ::: "memory");

    // reload P as an A-layout fragment (K = 32 j-columns)
    v16bf pf = afrag_bf16(pl + (size_t)n * 32, 0, khalf);

    // ---- O += P @ V : four 16x16 tiles over the 64 head dims -------------
#pragma unroll
    for (int dc = 0; dc < 4; ++dc) {
      v16bf vf;
#pragma unroll
      for (int e = 0; e < 16; ++e)
        vf[e] = (bf16_t)Vt[((khalf << 4) + e) * 64 + dc * 16 + n];
      o[dc] = wmma_bf16(pf, vf, o[dc]);
    }
  }

  // ---- finalize: divide by row sums, write [s*B+b, h*64+d] ---------------
#pragma unroll
  for (int r = 0; r < 8; ++r) {
    float invl = 1.0f / l_run[r];
    int srow = qbase + r + (khalf << 3);
    size_t rb = ((size_t)srow * Bn + b) * ld + colh;
#pragma unroll
    for (int dc = 0; dc < 4; ++dc)
      Op[rb + dc * 16 + n] = o[dc][r] * invl;
  }
}

// ---------------------------------------------------------------------------
// out = LayerNorm(a + c) * g + beta ; optionally also emit bf16 copy.
// One 512-wide row per 128-thread block (4 f32 per thread).
// ---------------------------------------------------------------------------
__global__ void __launch_bounds__(128)
add_ln_kernel(const float* __restrict__ a, const float* __restrict__ c,
              const float* __restrict__ g, const float* __restrict__ beta,
              float* __restrict__ out, bf16_t* __restrict__ outb, int D) {
  int row = blockIdx.x;
  int t = threadIdx.x;
  size_t base = (size_t)row * D;
  int i0 = t * 4;

  float4 av = *(const float4*)(a + base + i0);
  float4 cv = *(const float4*)(c + base + i0);
  float v[4] = {av.x + cv.x, av.y + cv.y, av.z + cv.z, av.w + cv.w};

  float s = v[0] + v[1] + v[2] + v[3];
  float q = v[0] * v[0] + v[1] * v[1] + v[2] * v[2] + v[3] * v[3];
#pragma unroll
  for (int m = 16; m >= 1; m >>= 1) {
    s += __shfl_xor(s, m, 32);
    q += __shfl_xor(q, m, 32);
  }
  __shared__ float ss[4], qq[4];
  int wid = t >> 5, lane = t & 31;
  if (lane == 0) { ss[wid] = s; qq[wid] = q; }
  __syncthreads();
  float ts = ss[0] + ss[1] + ss[2] + ss[3];
  float tq = qq[0] + qq[1] + qq[2] + qq[3];
  float mu = ts / (float)D;
  float var = tq / (float)D - mu * mu;
  float rstd = rsqrtf(var + 1e-5f);
#pragma unroll
  for (int k = 0; k < 4; ++k) {
    float ov = (v[k] - mu) * rstd * g[i0 + k] + beta[i0 + k];
    out[base + i0 + k] = ov;
    if (outb) outb[base + i0 + k] = (bf16_t)ov;
  }
}

// ---------------------------------------------------------------------------
// Host orchestration
// ---------------------------------------------------------------------------
extern "C" void kernel_launch(void* const* d_in, const int* in_sizes, int n_in,
                              void* d_out, int out_size, void* d_ws, size_t ws_size,
                              hipStream_t stream) {
  (void)in_sizes; (void)n_in; (void)out_size; (void)ws_size;
  const int S = 2048, B = 2, D = 512, H = 8, FF = 2048, Wb = 5;
  const size_t M = (size_t)S * B;  // 4096 rows

  const float* x   = (const float*)d_in[0];
  const float* enc = (const float*)d_in[1];
  const float *saW[4], *saB[4], *caW[4], *caB[4];
  for (int i = 0; i < 4; ++i) { saW[i] = (const float*)d_in[2 + i];  saB[i] = (const float*)d_in[6 + i]; }
  for (int i = 0; i < 4; ++i) { caW[i] = (const float*)d_in[10 + i]; caB[i] = (const float*)d_in[14 + i]; }
  const float* W1  = (const float*)d_in[18];
  const float* W2  = (const float*)d_in[19];
  const float* b1  = (const float*)d_in[20];
  const float* b2  = (const float*)d_in[21];
  const float* lng[3] = {(const float*)d_in[22], (const float*)d_in[24], (const float*)d_in[26]};
  const float* lnb[3] = {(const float*)d_in[23], (const float*)d_in[25], (const float*)d_in[27]};

  char* base = (char*)d_ws;
  size_t off = 0;
  auto alloc = [&](size_t bytes) -> char* {
    off = (off + 255) & ~(size_t)255;
    char* p = base + off;
    off += bytes;
    return p;
  };

  bf16_t* xb    = (bf16_t*)alloc(M * D * 2);
  bf16_t* encb  = (bf16_t*)alloc(M * D * 2);
  bf16_t* wbuf[8];
  for (int i = 0; i < 8; ++i) wbuf[i] = (bf16_t*)alloc((size_t)D * D * 2);
  bf16_t* w1b   = (bf16_t*)alloc((size_t)FF * D * 2);
  bf16_t* w2b   = (bf16_t*)alloc((size_t)D * FF * 2);
  float*  Qf    = (float*)alloc(M * D * 4);
  float*  Kf    = (float*)alloc(M * D * 4);
  float*  Vf    = (float*)alloc(M * D * 4);
  float*  attn  = (float*)alloc(M * D * 4);
  bf16_t* attnb = (bf16_t*)alloc(M * D * 2);
  float*  proj  = (float*)alloc(M * D * 4);
  float*  x1    = (float*)alloc(M * D * 4);
  bf16_t* x1b   = (bf16_t*)alloc(M * D * 2);
  float*  x2    = (float*)alloc(M * D * 4);
  bf16_t* x2b   = (bf16_t*)alloc(M * D * 2);
  float*  hf    = (float*)alloc(M * FF * 4);
  bf16_t* hb    = (bf16_t*)alloc(M * FF * 2);

  auto cvt = [&](const float* src, bf16_t* dst, size_t nelem) {
    cvt_f32_bf16_kernel<<<dim3(512), dim3(256), 0, stream>>>(src, dst, nelem);
  };
  auto gemm = [&](const bf16_t* A, const bf16_t* Wt, const float* bias, float* C,
                  int Mm, int Nn, int Kk, int relu) {
    gemm_bf16_kernel<<<dim3(Nn / 128, Mm / 64), dim3(256), 0, stream>>>(
        A, Wt, bias, C, Mm, Nn, Kk, relu);
  };
  auto flash = [&](const float* Q, const float* K, const float* V, float* O) {
    flash_mask_attn_kernel<<<dim3((S / 16) / 4, B * H), dim3(128), 0, stream>>>(
        Q, K, V, O, S, B, H, Wb);
  };

  // convert operands once per call
  cvt(x, xb, M * D);
  cvt(enc, encb, M * D);
  for (int i = 0; i < 4; ++i) cvt(saW[i], wbuf[i], (size_t)D * D);
  for (int i = 0; i < 4; ++i) cvt(caW[i], wbuf[4 + i], (size_t)D * D);
  cvt(W1, w1b, (size_t)FF * D);
  cvt(W2, w2b, (size_t)D * FF);

  // ---- self attention ----
  gemm(xb, wbuf[0], saB[0], Qf, (int)M, D, D, 0);
  gemm(xb, wbuf[1], saB[1], Kf, (int)M, D, D, 0);
  gemm(xb, wbuf[2], saB[2], Vf, (int)M, D, D, 0);
  flash(Qf, Kf, Vf, attn);
  cvt(attn, attnb, M * D);
  gemm(attnb, wbuf[3], saB[3], proj, (int)M, D, D, 0);
  add_ln_kernel<<<dim3((unsigned)M), dim3(128), 0, stream>>>(x, proj, lng[0], lnb[0], x1, x1b, D);

  // ---- cross attention ----
  gemm(x1b,  wbuf[4], caB[0], Qf, (int)M, D, D, 0);
  gemm(encb, wbuf[5], caB[1], Kf, (int)M, D, D, 0);
  gemm(encb, wbuf[6], caB[2], Vf, (int)M, D, D, 0);
  flash(Qf, Kf, Vf, attn);
  cvt(attn, attnb, M * D);
  gemm(attnb, wbuf[7], caB[3], proj, (int)M, D, D, 0);
  add_ln_kernel<<<dim3((unsigned)M), dim3(128), 0, stream>>>(x1, proj, lng[1], lnb[1], x2, x2b, D);

  // ---- feed-forward ----
  gemm(x2b, w1b, b1, hf, (int)M, FF, D, 1);
  cvt(hf, hb, M * FF);
  gemm(hb, w2b, b2, proj, (int)M, D, FF, 0);
  add_ln_kernel<<<dim3((unsigned)M), dim3(128), 0, stream>>>(x2, proj, lng[2], lnb[2],
                                                             (float*)d_out, (bf16_t*)nullptr, D);
}
